// LAINetOriginal_51900384805146
// MI455X (gfx1250) — matrix operand order repaired and checked by reading, not compiled, not run on hardware
//
#include <hip/hip_runtime.h>
#include <hip/hip_bf16.h>
#include <math.h>

// ---------------- problem constants ----------------
#define BATCH   64
#define NWIN    1000
#define WIN     500
#define HID     30
#define ANC     7
#define KSZ     75
#define PADH    37          // KSZ/2
#define OB_ELEMS (BATCH*ANC*NWIN*2)   // 896000

// WMMA fragment types (probe-verified)
typedef __attribute__((ext_vector_type(16))) _Float16 v16h;
typedef __attribute__((ext_vector_type(8)))  float    v8f;
typedef __attribute__((ext_vector_type(2)))  _Float16 v2h;

// LDS pitches (halfwords / floats) padded against bank conflicts
#define AP 136   // A tile pitch (f16), rows = 128, K chunk = 128
#define BP 136   // B tile pitch (f16), cols = 32 (N padded), K chunk = 128
#define HP 33    // h tile pitch (f32), 128 x 32

// =====================================================================
// Kernel A: per-window fused  GEMM(128x500x30, f16 WMMA) -> +b1 -> ReLU
//           -> BatchNorm over batch axis -> GEMM(128x30x7) -> +b2
//           -> write out_base[b,a,n,c]
// one block per window n; 256 threads = 8 wave32
// =====================================================================
__global__ __launch_bounds__(256)
void fused_mlp_bn_kernel(const float* __restrict__ x,
                         const float* __restrict__ W1,
                         const float* __restrict__ b1,
                         const float* __restrict__ W2,
                         const float* __restrict__ b2,
                         float* __restrict__ out_base)
{
    __shared__ _Float16 As[128 * AP];      // A chunk, rows r = c*64+b
    __shared__ _Float16 Bs[32 * BP];       // W1 chunk, stored [col][k]
    __shared__ float    hs[128 * HP];      // post-ReLU hidden tile
    __shared__ float    meanS[2][32];
    __shared__ float    rstdS[2][32];

    const int n    = blockIdx.x;
    const int t    = threadIdx.x;
    const int wave = t >> 5;
    const int lane = t & 31;
    const int mrow = lane & 15;
    const int half = lane >> 4;

    v8f acc0 = {0.f,0.f,0.f,0.f,0.f,0.f,0.f,0.f};
    v8f acc1 = {0.f,0.f,0.f,0.f,0.f,0.f,0.f,0.f};

    // ---- K loop: 4 chunks of 128 (K padded 500 -> 512; B cols zeroed) ----
    for (int ch = 0; ch < 4; ++ch) {
        const int w0 = ch * 128;

        // stage A: x[b, n*500+w, c] -> f16, rows r = c*64+b
        #pragma unroll
        for (int it = 0; it < 16; ++it) {
            int idx = t + 256 * it;          // 0..4095
            int b   = idx >> 6;              // 0..63
            int q   = idx & 63;              // 0..63
            int wl  = q * 2;                 // local k (covers wl, wl+1)
            int w   = w0 + wl;
            float e0 = 0.f, e1 = 0.f, e2 = 0.f, e3 = 0.f;
            if (w + 1 < WIN) {
                const float4 v = *(const float4*)(x + (size_t)b*1000000
                                                    + (size_t)n*1000 + (size_t)w*2);
                e0 = v.x; e1 = v.y; e2 = v.z; e3 = v.w;
            } else if (w < WIN) {
                const float* p = x + (size_t)b*1000000 + (size_t)n*1000 + (size_t)w*2;
                e0 = p[0]; e1 = p[1];
            }
            // (x-0.5)*2, convert to f16, packed 2xf16 stores
            v2h pc0 = { (_Float16)((e0 - 0.5f)*2.0f), (_Float16)((e2 - 0.5f)*2.0f) };
            v2h pc1 = { (_Float16)((e1 - 0.5f)*2.0f), (_Float16)((e3 - 0.5f)*2.0f) };
            *(v2h*)&As[ b       * AP + wl] = pc0;   // c=0 row
            *(v2h*)&As[(64 + b) * AP + wl] = pc1;   // c=1 row
        }
        // stage B: W1[n, w0..w0+127, 0..29] transposed into Bs[col][k]
        #pragma unroll
        for (int it = 0; it < 15; ++it) {
            int idx = t + 256 * it;          // 0..3839 = 128*30
            int kl  = idx / HID;
            int hc  = idx % HID;
            int w   = w0 + kl;
            float v = (w < WIN) ? W1[(size_t)n*15000 + (size_t)w*HID + hc] : 0.0f;
            Bs[hc * BP + kl] = (_Float16)v;
        }
        // zero the two padded N columns (30, 31)
        { int col = 30 + (t >> 7); int kl = t & 127; Bs[col * BP + kl] = (_Float16)0.0f; }
        __syncthreads();

        // WMMA: wave owns M-tile `wave`, both N-tiles. 4 k-steps of 32.
        const int r = wave * 16 + mrow;      // A: M = lane%16
        #pragma unroll
        for (int ks = 0; ks < 4; ++ks) {
            const int kb = ks * 32;
            v16h a, bv0, bv1;
            // A frag: lanes<16 K=0..15 ; lanes>=16 K=16..31
            const _Float16* ap = &As[r * AP + kb + half * 16];
            #pragma unroll
            for (int j = 0; j < 16; ++j) { a[j] = ap[j]; }
            // B frag: lane holds column N=lane%16, K = half*16 + 0..15
            const _Float16* bp0 = &Bs[ mrow       * BP + kb + half * 16];
            const _Float16* bp1 = &Bs[(16 + mrow) * BP + kb + half * 16];
            #pragma unroll
            for (int j = 0; j < 16; ++j) { bv0[j] = bp0[j]; bv1[j] = bp1[j]; }
            acc0 = __builtin_amdgcn_wmma_f32_16x16x32_f16(false, a, false, bv0,
                                                          (short)0, acc0, false, false);
            acc1 = __builtin_amdgcn_wmma_f32_16x16x32_f16(false, a, false, bv1,
                                                          (short)0, acc1, false, false);
        }
        __syncthreads();
    }

    // ---- epilogue: +b1, ReLU, spill h tile to LDS ----
    {
        const int colA = mrow;
        const int colB = 16 + mrow;
        const float biasA = (colA < HID) ? b1[n * HID + colA] : 0.0f;
        const float biasB = (colB < HID) ? b1[n * HID + colB] : 0.0f;
        #pragma unroll
        for (int i = 0; i < 8; ++i) {
            int rloc = wave * 16 + half * 8 + i;    // D: M = vgpr + 8*(lane/16)
            float vA = acc0[i] + biasA; vA = vA > 0.f ? vA : 0.f;
            float vB = acc1[i] + biasB; vB = vB > 0.f ? vB : 0.f;
            hs[rloc * HP + colA] = vA;
            hs[rloc * HP + colB] = vB;
        }
    }
    __syncthreads();

    // ---- batch statistics over b (rows c*64 + 0..63) ----
    if (t < 2 * HID) {
        int c = t / HID, hh = t % HID;
        float s = 0.f, s2 = 0.f;
        #pragma unroll 8
        for (int b = 0; b < BATCH; ++b) {
            float v = hs[(c * 64 + b) * HP + hh];
            s += v; s2 += v * v;
        }
        float m   = s * (1.0f / BATCH);
        float var = s2 * (1.0f / BATCH) - m * m;
        meanS[c][hh] = m;
        rstdS[c][hh] = rsqrtf(var + 1e-5f);
    }
    __syncthreads();

    // ---- tiny GEMM2 (K=30, N=7) + b2, write out_base[b,a,n,c] ----
    #pragma unroll
    for (int q = 0; q < 4; ++q) {
        int idx = t + 256 * q;               // 0..1023, need < 896
        if (idx < 128 * ANC) {
            int r = idx / ANC, a = idx % ANC;
            int c = r >> 6,    b = r & 63;
            float acc = b2[n * ANC + a];
            #pragma unroll
            for (int hh = 0; hh < HID; ++hh) {
                float v = (hs[r * HP + hh] - meanS[c][hh]) * rstdS[c][hh];
                acc += v * W2[(size_t)n * (HID * ANC) + hh * ANC + a];
            }
            out_base[(size_t)b * 14000 + (size_t)a * 2000 + n * 2 + c] = acc;
        }
    }
}

// =====================================================================
// Kernel B: fused softmax + reflect-padded conv as WMMA GEMM.
//   For fixed b:  out[n,(o,c')] = sum_kk  A[n,kk] * W'[kk,(o,c')]
//   kk = anchor i (7) x tap k (75) x channel cc (2), per-anchor K padded
//   150 -> 160 so each 32-wide k-step stays inside one anchor slab:
//   A fragment = contiguous f16 reads from the softmaxed p-slab at
//   offset n_local*2 (im2col = pointer shift). Channel reflect-pad and
//   K-padding are folded into W' (zero weights kill padded lanes).
// one block: fixed b, 128 n values; 256 threads = 8 wave32
// =====================================================================
#define IBLK  160                 // padded per-anchor K
#define KP2   (ANC * IBLK)        // 1120 = 35 k-steps of 32
#define PLSZ  2848                // 7*404 = 2828 valid + zeroed slack

__global__ __launch_bounds__(256)
void softmax_conv_wmma_kernel(const float* __restrict__ ob,
                              const float* __restrict__ conv_w,
                              const float* __restrict__ conv_b,
                              float* __restrict__ outs)
{
    __shared__ _Float16 Pl[PLSZ];        // [i][ml*2+cc], f16 softmax probs
    __shared__ _Float16 Wl[16 * KP2];    // W' transposed: [col(16)][kk(1120)]

    const int b  = blockIdx.y;
    const int n0 = blockIdx.x * 128;
    const int t  = threadIdx.x;
    const int lane = t & 31;
    const int wave = t >> 5;
    const int mrow = lane & 15;
    const int half = lane >> 4;

    // ---- stage W' (with channel-swap + zero K-padding) ----
    for (int idx = t; idx < 16 * KP2; idx += 256) {
        int col = idx / KP2;             // (o,c') ; cols 14,15 = zero pad
        int kk  = idx % KP2;
        int i   = kk / IBLK;
        int off = kk % IBLK;
        int k   = off >> 1;
        int cc  = off & 1;
        float v = 0.f;
        if (col < 14 && k < KSZ) {
            int o  = col >> 1, cp = col & 1;
            int j  = cp ? cc : 1 - cc;   // reflect pad of width-2 channel axis
            v = conv_w[((o * ANC + i) * KSZ + k) * 2 + j];
        }
        Wl[idx] = (_Float16)v;
    }

    // ---- stage p-slab: softmax over 7 anchors, reflect in H ----
    for (int idx = t; idx < 202 * 2; idx += 256) {
        int ml = idx >> 1, cc = idx & 1;
        int m  = n0 - PADH + ml;         // reflect into [0, 999]
        if (m < 0)   m = -m;
        if (m > 999) m = 1998 - m;
        const float* src = ob + (size_t)b * 14000 + m * 2 + cc;
        float v[ANC], mx = -3.4e38f;
        #pragma unroll
        for (int a = 0; a < ANC; ++a) { v[a] = src[a * 2000]; mx = fmaxf(mx, v[a]); }
        float s = 0.f;
        #pragma unroll
        for (int a = 0; a < ANC; ++a) { v[a] = expf(v[a] - mx); s += v[a]; }
        float inv = 1.0f / s;
        #pragma unroll
        for (int a = 0; a < ANC; ++a) Pl[a * 404 + idx] = (_Float16)(v[a] * inv);
    }
    if (t < PLSZ - 2828) Pl[2828 + t] = (_Float16)0.0f;   // finite slack
    __syncthreads();

    // ---- WMMA: wave owns M-tile `wave` (16 n-values), 35 k-steps ----
    v8f acc = {0.f,0.f,0.f,0.f,0.f,0.f,0.f,0.f};
    const int nrow = wave * 16 + mrow;   // A: M = lane%16
    #pragma unroll
    for (int i = 0; i < ANC; ++i) {
        #pragma unroll
        for (int s = 0; s < 5; ++s) {
            v16h a, bv;
            // A frag: contiguous reads inside anchor i's slab (im2col shift)
            const _Float16* ap = &Pl[i * 404 + nrow * 2 + s * 32 + half * 16];
            #pragma unroll
            for (int j = 0; j < 16; ++j) { a[j] = ap[j]; }
            // B frag: lane holds column col=lane%16
            const _Float16* bp = &Wl[mrow * KP2 + i * IBLK + s * 32 + half * 16];
            #pragma unroll
            for (int j = 0; j < 16; ++j) { bv[j] = bp[j]; }
            acc = __builtin_amdgcn_wmma_f32_16x16x32_f16(false, a, false, bv,
                                                         (short)0, acc, false, false);
        }
    }

    // ---- epilogue: + conv_b, write out_smooth[b,o,n,c'] ----
    const int col = mrow;                // D: N = lane%16
    if (col < 14) {
        const int o  = col >> 1, cp = col & 1;
        const float bias = conv_b[o];
        #pragma unroll
        for (int i = 0; i < 8; ++i) {
            int nl = wave * 16 + half * 8 + i;   // D: M = vgpr + 8*(lane/16)
            int nn = n0 + nl;
            if (nn < NWIN)
                outs[(size_t)b * 14000 + (size_t)o * 2000 + nn * 2 + cp] = acc[i] + bias;
        }
    }
}

// =====================================================================
extern "C" void kernel_launch(void* const* d_in, const int* in_sizes, int n_in,
                              void* d_out, int out_size, void* d_ws, size_t ws_size,
                              hipStream_t stream)
{
    const float* x      = (const float*)d_in[0];
    const float* W1     = (const float*)d_in[1];
    const float* b1     = (const float*)d_in[2];
    const float* W2     = (const float*)d_in[3];
    const float* b2     = (const float*)d_in[4];
    const float* conv_w = (const float*)d_in[5];
    const float* conv_b = (const float*)d_in[6];

    float* out_base   = (float*)d_out;            // 896000 floats
    float* out_smooth = (float*)d_out + OB_ELEMS; // 896000 floats
    (void)d_ws; (void)ws_size; (void)in_sizes; (void)n_in; (void)out_size;

    fused_mlp_bn_kernel<<<NWIN, 256, 0, stream>>>(x, W1, b1, W2, b2, out_base);
    softmax_conv_wmma_kernel<<<dim3(8, BATCH), 256, 0, stream>>>(out_base, conv_w,
                                                                 conv_b, out_smooth);
}
